// TransformerStack_28965259444254
// MI455X (gfx1250) — matrix operand order repaired and checked by reading, not compiled, not run on hardware
//
#include <hip/hip_runtime.h>
#include <hip/hip_bf16.h>
#include <math.h>

#define B_ 2
#define L_ 1024
#define D_ 768
#define H_ 12
#define DH_ 64
#define DP_ 32
#define MLP_ 3072
#define NL_ 4
#define WIN_ 128
#define BAND_ (2 * WIN_ + 1) /* 257 */

typedef __bf16 bf16_t;
typedef __attribute__((ext_vector_type(16))) __bf16 v16bf;
typedef __attribute__((ext_vector_type(8)))  __bf16 v8bf;
typedef __attribute__((ext_vector_type(8)))  float  v8f;

union V16U { v16bf v; v8bf h[2]; };

__device__ __forceinline__ bf16_t f2bf(float f) {
  unsigned u = __float_as_uint(f);
  u += 0x7FFFu + ((u >> 16) & 1u);           // round-to-nearest-even
  unsigned short hs = (unsigned short)(u >> 16);
  return __builtin_bit_cast(bf16_t, hs);
}

// ---- CDNA5 async global->LDS staging (ASYNCcnt path), with safe fallback ---
#if __has_builtin(__builtin_amdgcn_global_load_async_to_lds_b128)
#define HAVE_ASYNC_LDS 1
typedef int gv4i __attribute__((vector_size(16)));
typedef __attribute__((address_space(1))) gv4i as1_gv4i;
typedef __attribute__((address_space(3))) gv4i as3_gv4i;
#endif

__device__ __forceinline__ void async_cp16(const bf16_t* g, bf16_t* l) {
#if defined(HAVE_ASYNC_LDS)
  __builtin_amdgcn_global_load_async_to_lds_b128((as1_gv4i*)g, (as3_gv4i*)l, 0, 0);
#else
  *(v8bf*)l = *(const v8bf*)g;   // load + ds_store fallback
#endif
}

__device__ __forceinline__ void wait_async() {
#if defined(HAVE_ASYNC_LDS)
#if __has_builtin(__builtin_amdgcn_s_wait_asynccnt)
  __builtin_amdgcn_s_wait_asynccnt(0);
#else
  asm volatile("s_wait_asynccnt 0x0" ::: "memory");
#endif
#endif
}

// ---------------- pair bias on the window band: LN(DP) + dot(pb_w) ----------
__global__ __launch_bounds__(256) void pair_bias_kernel(
    const float* __restrict__ pair, const float* __restrict__ ln_g,
    const float* __restrict__ ln_b, const float* __restrict__ pb_w,
    const float* __restrict__ pb_b_all, int layer, float* __restrict__ band)
{
  int idx = blockIdx.x * 256 + threadIdx.x;
  int total = B_ * L_ * BAND_;
  if (idx >= total) return;
  int jw = idx % BAND_;
  int rest = idx / BAND_;
  int i = rest % L_;
  int b = rest / L_;
  int j = i - WIN_ + jw;
  float out = -1e30f;
  if (j >= 0 && j < L_) {
    const float* p = pair + (((size_t)b * L_ + i) * L_ + j) * DP_;
    float v[DP_];
    float mean = 0.f;
#pragma unroll
    for (int t = 0; t < DP_; ++t) { v[t] = p[t]; mean += v[t]; }
    mean *= (1.0f / DP_);
    float var = 0.f;
#pragma unroll
    for (int t = 0; t < DP_; ++t) { float d = v[t] - mean; var += d * d; }
    var *= (1.0f / DP_);
    float inv = rsqrtf(var + 1e-5f);
    float acc = pb_b_all[layer];
#pragma unroll
    for (int t = 0; t < DP_; ++t)
      acc += ((v[t] - mean) * inv * ln_g[t] + ln_b[t]) * pb_w[t];
    out = acc;
  }
  band[idx] = out;
}

// ---------------- rmsnorm (fp32) + cast to bf16 -----------------------------
__global__ __launch_bounds__(256) void rmsnorm_cast_kernel(
    const float* __restrict__ s, bf16_t* __restrict__ h)
{
  int row = blockIdx.x;
  const float* x = s + (size_t)row * D_;
  __shared__ float red[256];
  float ss = 0.f;
  for (int i = threadIdx.x; i < D_; i += 256) { float v = x[i]; ss += v * v; }
  red[threadIdx.x] = ss;
  __syncthreads();
  for (int ofs = 128; ofs > 0; ofs >>= 1) {
    if ((int)threadIdx.x < ofs) red[threadIdx.x] += red[threadIdx.x + ofs];
    __syncthreads();
  }
  float inv = rsqrtf(red[0] * (1.0f / D_) + 1.1920929e-07f);
  for (int i = threadIdx.x; i < D_; i += 256)
    h[(size_t)row * D_ + i] = f2bf(x[i] * inv);
}

// ---------------- fp32 -> bf16 weight conversion ----------------------------
__global__ __launch_bounds__(256) void w2bf_kernel(const float* __restrict__ w,
                                                   bf16_t* __restrict__ o, int n)
{
  int i = blockIdx.x * 256 + threadIdx.x;
  if (i < n) o[i] = f2bf(w[i]);
}

// ---------------- WMMA bf16 GEMM: C[M,N] = A[M,K] * W[N,K]^T + bias ---------
// Block tile 64(M) x 128(N), 8 waves (2x4) of 32x32 wave-tiles, K-step 32.
// Double-buffered async global->LDS staging.
// MODE 0: Cf = acc+bias   MODE 1: Cf += acc+bias   MODE 2: Cb = gelu(acc+bias)
template <int MODE>
__global__ __launch_bounds__(256) void gemm_bf16_kernel(
    const bf16_t* __restrict__ A, const bf16_t* __restrict__ W,
    const float* __restrict__ bias, float* __restrict__ Cf,
    bf16_t* __restrict__ Cb, int M, int N, int K)
{
  __shared__ __align__(128) bf16_t At[2][64 * 32];
  __shared__ __align__(128) bf16_t Bt[2][128 * 32];

  int tid = threadIdx.x;
  int waveId = tid >> 5;
  int lane = tid & 31;
  int r = lane & 15, hi = lane >> 4;

  int bm = blockIdx.x % (M >> 6);
  int bn = blockIdx.x / (M >> 6);
  int m0 = bm * 64;
  int n0 = bn * 128;

  const bf16_t* Abase = A + (size_t)m0 * K;
  const bf16_t* Wbase = W + (size_t)n0 * K;

  // cooperative staging coordinates: 16B (8 bf16) chunks
  int arow = tid >> 2, acol = (tid & 3) * 8;          // A: 256 chunks, 1/thread
  int bc0 = tid, bc1 = tid + 256;                      // B: 512 chunks, 2/thread
  int br0 = bc0 >> 2, bk0c = (bc0 & 3) * 8;
  int br1 = bc1 >> 2, bk1c = (bc1 & 3) * 8;

  int wm = waveId >> 2;        // 0..1  (M sub-tile)
  int wn = waveId & 3;         // 0..3  (N sub-tile)
  int aoff = hi * 16;          // A frag: contiguous K chunk {aoff..aoff+15}
  int boff = hi * 16;          // B frag: contiguous K chunk {boff..boff+15}

  auto stage = [&](int buf, int k0) {
    async_cp16(Abase + (size_t)arow * K + k0 + acol, &At[buf][arow * 32 + acol]);
    async_cp16(Wbase + (size_t)br0 * K + k0 + bk0c, &Bt[buf][br0 * 32 + bk0c]);
    async_cp16(Wbase + (size_t)br1 * K + k0 + bk1c, &Bt[buf][br1 * 32 + bk1c]);
    __builtin_prefetch(Abase + (size_t)arow * K + k0 + acol + 32, 0, 1);
    __builtin_prefetch(Wbase + (size_t)br0 * K + k0 + bk0c + 32, 0, 1);
  };

  v8f c[2][2] = {{{}, {}}, {{}, {}}};

  stage(0, 0);
  wait_async();
  __syncthreads();

  int cur = 0;
  for (int k0 = 0; k0 < K; k0 += 32) {
    if (k0 + 32 < K) stage(cur ^ 1, k0 + 32);   // overlap next tile with WMMA

    const bf16_t* la0 = &At[cur][(wm * 32 + r) * 32];
    const bf16_t* la1 = &At[cur][(wm * 32 + 16 + r) * 32];
    const bf16_t* lb0 = &Bt[cur][(wn * 32 + r) * 32];
    const bf16_t* lb1 = &Bt[cur][(wn * 32 + 16 + r) * 32];

    V16U a[2];
    v16bf bmat[2];
    a[0].h[0] = *(const v8bf*)(la0 + aoff);
    a[0].h[1] = *(const v8bf*)(la0 + aoff + 8);
    a[1].h[0] = *(const v8bf*)(la1 + aoff);
    a[1].h[1] = *(const v8bf*)(la1 + aoff + 8);
    bmat[0] = *(const v16bf*)(lb0 + boff);
    bmat[1] = *(const v16bf*)(lb1 + boff);

#pragma unroll
    for (int x = 0; x < 2; ++x)
#pragma unroll
      for (int y = 0; y < 2; ++y)
        c[x][y] = __builtin_amdgcn_wmma_f32_16x16x32_bf16(
            false, a[x].v, false, bmat[y], (short)0, c[x][y], false, false);

    wait_async();      // next-tile async copies complete (own wave's)
    __syncthreads();   // all waves' copies visible + done reading cur
    cur ^= 1;
  }

#pragma unroll
  for (int x = 0; x < 2; ++x) {
#pragma unroll
    for (int y = 0; y < 2; ++y) {
      int nn = n0 + wn * 32 + y * 16 + r;
      float bv = bias[nn];
#pragma unroll
      for (int v = 0; v < 8; ++v) {
        int m = m0 + wm * 32 + x * 16 + hi * 8 + v;
        float val = c[x][y][v] + bv;
        size_t o = (size_t)m * N + nn;
        if (MODE == 0) {
          Cf[o] = val;
        } else if (MODE == 1) {
          Cf[o] += val;
        } else {
          float g = 0.5f * val * (1.0f + erff(val * 0.70710678118654752f));
          Cb[o] = f2bf(g);
        }
      }
    }
  }
}

// ---------------- RoPE on q,k inside the qkv buffer -------------------------
__global__ __launch_bounds__(256) void rope_kernel(float* __restrict__ qkv)
{
  int idx = blockIdx.x * 256 + threadIdx.x;
  int total = B_ * L_ * H_ * (DH_ / 2);
  if (idx >= total) return;
  int d = idx % (DH_ / 2);
  int rest = idx / (DH_ / 2);
  int h = rest % H_;
  int row = rest / H_;  // b*L + l
  int l = row % L_;
  float inv_freq = powf(10000.0f, -(float)d * (1.0f / 32.0f));
  float ang = (float)l * inv_freq;
  float cs = cosf(ang), sn = sinf(ang);
  float* base = qkv + (size_t)row * (3 * D_) + h * DH_;
  float q1 = base[d], q2 = base[d + 32];
  base[d] = q1 * cs - q2 * sn;
  base[d + 32] = q2 * cs + q1 * sn;
  float* kb = base + D_;
  float k1 = kb[d], k2 = kb[d + 32];
  kb[d] = k1 * cs - k2 * sn;
  kb[d + 32] = k2 * cs + k1 * sn;
}

// ---------------- windowed attention: one 64-thread block per (b,h,q) -------
__global__ __launch_bounds__(64) void attn_kernel(
    const float* __restrict__ qkv, const float* __restrict__ band,
    bf16_t* __restrict__ y)
{
  int blk = blockIdx.x;  // ((b*H + h)*L + i)
  int i = blk % L_;
  int bh = blk / L_;
  int h = bh % H_;
  int b = bh / H_;
  int d = threadIdx.x;  // 0..63
  int row = b * L_ + i;

  __shared__ float qsh[DH_];
  __shared__ float sc[BAND_];
  __shared__ float red[64];

  const float* qrow = qkv + (size_t)row * (3 * D_) + h * DH_;
  qsh[d] = qrow[d] * 0.125f;  // 1/sqrt(64)
  __syncthreads();

  int j0 = i - WIN_; if (j0 < 0) j0 = 0;
  int j1 = i + WIN_; if (j1 > L_ - 1) j1 = L_ - 1;
  int nj = j1 - j0 + 1;

  for (int jj = d; jj < nj; jj += 64) {
    int j = j0 + jj;
    const float* krow = qkv + (size_t)(b * L_ + j) * (3 * D_) + D_ + h * DH_;
    float acc = 0.f;
#pragma unroll
    for (int t = 0; t < DH_; ++t) acc += qsh[t] * krow[t];
    sc[jj] = acc + band[(size_t)row * BAND_ + (j - i + WIN_)];
  }
  __syncthreads();

  float m = -1e30f;
  for (int jj = d; jj < nj; jj += 64) m = fmaxf(m, sc[jj]);
  red[d] = m;
  __syncthreads();
  for (int ofs = 32; ofs > 0; ofs >>= 1) {
    if (d < ofs) red[d] = fmaxf(red[d], red[d + ofs]);
    __syncthreads();
  }
  float mx = red[0];
  __syncthreads();

  float ssum = 0.f;
  for (int jj = d; jj < nj; jj += 64) {
    float e = __expf(sc[jj] - mx);
    sc[jj] = e;
    ssum += e;
  }
  red[d] = ssum;
  __syncthreads();
  for (int ofs = 32; ofs > 0; ofs >>= 1) {
    if (d < ofs) red[d] += red[d + ofs];
    __syncthreads();
  }
  float denom = red[0];

  float acc = 0.f;
  const float* vbase = qkv + (size_t)(b * L_ + j0) * (3 * D_) + 2 * D_ + h * DH_ + d;
  for (int jj = 0; jj < nj; ++jj)
    acc += sc[jj] * vbase[(size_t)jj * (3 * D_)];
  y[(size_t)row * D_ + h * DH_ + d] = f2bf(acc / denom);
}

// ---------------------------------------------------------------------------
extern "C" void kernel_launch(void* const* d_in, const int* in_sizes, int n_in,
                              void* d_out, int out_size, void* d_ws, size_t ws_size,
                              hipStream_t stream)
{
  const float* s_in  = (const float*)d_in[0];
  const float* pair  = (const float*)d_in[1];
  const float* qkv_w = (const float*)d_in[2];
  const float* qkv_b = (const float*)d_in[3];
  const float* out_w = (const float*)d_in[4];
  const float* out_b = (const float*)d_in[5];
  const float* fc1_w = (const float*)d_in[6];
  const float* fc1_b = (const float*)d_in[7];
  const float* fc2_w = (const float*)d_in[8];
  const float* fc2_b = (const float*)d_in[9];
  const float* ln_g  = (const float*)d_in[10];
  const float* ln_b  = (const float*)d_in[11];
  const float* pb_w  = (const float*)d_in[12];
  const float* pb_b  = (const float*)d_in[13];
  float* s = (float*)d_out;  // residual stream lives in d_out

  const int ROWS = B_ * L_;
  char* wsp = (char*)d_ws;
  auto carve = [&](size_t bytes) -> char* {
    char* p = wsp;
    wsp += (bytes + 255) & ~(size_t)255;
    return p;
  };
  float*  qkvbuf = (float*) carve((size_t)ROWS * 3 * D_ * 4);  // 18.9 MB
  bf16_t* hbf    = (bf16_t*)carve((size_t)ROWS * D_ * 2);      //  3.1 MB
  bf16_t* gbf    = (bf16_t*)carve((size_t)ROWS * MLP_ * 2);    // 12.6 MB
  bf16_t* ybf    = (bf16_t*)carve((size_t)ROWS * D_ * 2);      //  3.1 MB
  float*  band   = (float*) carve((size_t)ROWS * BAND_ * 4);   //  2.1 MB
  bf16_t* wbf    = (bf16_t*)carve((size_t)MLP_ * D_ * 2);      //  4.7 MB

  (void)hipMemcpyAsync(s, s_in, (size_t)ROWS * D_ * 4, hipMemcpyDeviceToDevice, stream);

  for (int l = 0; l < NL_; ++l) {
    {
      int total = B_ * L_ * BAND_;
      pair_bias_kernel<<<(total + 255) / 256, 256, 0, stream>>>(
          pair, ln_g + l * DP_, ln_b + l * DP_, pb_w + l * DP_, pb_b, l, band);
    }
    rmsnorm_cast_kernel<<<ROWS, 256, 0, stream>>>(s, hbf);
    {
      int n = 3 * D_ * D_;
      w2bf_kernel<<<(n + 255) / 256, 256, 0, stream>>>(qkv_w + (size_t)l * n, wbf, n);
      int blocks = (ROWS >> 6) * ((3 * D_) >> 7);
      gemm_bf16_kernel<0><<<blocks, 256, 0, stream>>>(
          hbf, wbf, qkv_b + l * 3 * D_, qkvbuf, (bf16_t*)nullptr, ROWS, 3 * D_, D_);
    }
    {
      int total = B_ * L_ * H_ * (DH_ / 2);
      rope_kernel<<<(total + 255) / 256, 256, 0, stream>>>(qkvbuf);
    }
    attn_kernel<<<B_ * H_ * L_, 64, 0, stream>>>(qkvbuf, band, ybf);
    {
      int n = D_ * D_;
      w2bf_kernel<<<(n + 255) / 256, 256, 0, stream>>>(out_w + (size_t)l * n, wbf, n);
      int blocks = (ROWS >> 6) * (D_ >> 7);
      gemm_bf16_kernel<1><<<blocks, 256, 0, stream>>>(
          ybf, wbf, out_b + l * D_, s, (bf16_t*)nullptr, ROWS, D_, D_);
    }
    rmsnorm_cast_kernel<<<ROWS, 256, 0, stream>>>(s, hbf);
    {
      int n = MLP_ * D_;
      w2bf_kernel<<<(n + 255) / 256, 256, 0, stream>>>(fc1_w + (size_t)l * n, wbf, n);
      int blocks = (ROWS >> 6) * (MLP_ >> 7);
      gemm_bf16_kernel<2><<<blocks, 256, 0, stream>>>(
          hbf, wbf, fc1_b + l * MLP_, (float*)nullptr, gbf, ROWS, MLP_, D_);
    }
    {
      int n = D_ * MLP_;
      w2bf_kernel<<<(n + 255) / 256, 256, 0, stream>>>(fc2_w + (size_t)l * n, wbf, n);
      int blocks = (ROWS >> 6) * (D_ >> 7);
      gemm_bf16_kernel<1><<<blocks, 256, 0, stream>>>(
          gbf, wbf, fc2_b + l * D_, s, (bf16_t*)nullptr, ROWS, D_, MLP_);
    }
  }
  (void)in_sizes; (void)n_in; (void)out_size; (void)ws_size;
}
